// DMPNNEncoder_50672024158351
// MI455X (gfx1250) — compile-verified, matrix operand order checked by reading, not run on hardware
//
#include <hip/hip_runtime.h>

// ---------------- problem constants (from reference) ----------------
constexpr int kAtoms  = 100000;   // multiple of 16 -> 6250 tiles
constexpr int kBonds  = 200000;   // multiple of 16 -> 12500 tiles
constexpr int kMaxNb  = 6;
constexpr int kGraphs = 4096;
constexpr int kAtomF  = 133;
constexpr int kBondF  = 14;
constexpr int kHidden = 300;
constexpr int kDepth  = 8;

// padded GEMM dims (multiples of WMMA 16x16x32 bf16 shape)
constexpr int HP      = 320;          // hidden padded (20 n-tiles, 10 k-steps)
constexpr int NT      = HP / 16;      // 20
constexpr int KH_KT   = HP / 32;      // 10
constexpr int KIN     = 160;          // 147 -> 160 (5 k-steps)
constexpr int KIN_KT  = KIN / 32;
constexpr int KOUT    = 448;          // 433 -> 448 (14 k-steps)
constexpr int KOUT_KT = KOUT / 32;

// LDS row strides in ushort units; row*stride*2 bytes is a multiple of 16,
// and stride avoids power-of-two bank collisions (328*2/4 = 164 dwords, 164%64=36)
constexpr int S_IN  = 168;   // input tile  16 x 160
constexpr int S_NEI = 328;   // message tile 16 x 320
constexpr int S_OUT = 456;   // readout tile 16 x 448

typedef __attribute__((ext_vector_type(16))) __bf16 bf16x16;
typedef __attribute__((ext_vector_type(8)))  float  f32x8;

struct alignas(16) U128 { unsigned int x, y, z, w; };
union B128 { U128 u; unsigned short s[8]; };

struct alignas(32) Frag {
  union { bf16x16 b; unsigned short s[16]; U128 q[2]; };
};

__device__ __forceinline__ unsigned short f32_to_bf16(float f) {
  unsigned u = __builtin_bit_cast(unsigned, f);
  unsigned r = u + 0x7FFFu + ((u >> 16) & 1u);   // round-to-nearest-even
  return (unsigned short)(r >> 16);
}
__device__ __forceinline__ float bf16_to_f32(unsigned short h) {
  unsigned u = ((unsigned)h) << 16;
  return __builtin_bit_cast(float, u);
}

__device__ __forceinline__ f32x8 wmma_bf16(const Frag& a, const Frag& b, f32x8 c) {
  // D = A(16x32 bf16) x B(32x16 bf16) + C(16x16 f32)
  return __builtin_amdgcn_wmma_f32_16x16x32_bf16(
      false, a.b, false, b.b, (short)0, c, false, false);
}

// A fragment from LDS tile (16 rows x stride), ISA 16-bit A layout:
// element e of lane L: M = L%16, K = kt*32 + (e/8)*16 + 8*(L/16) + (e%8)
__device__ __forceinline__ void load_a_lds(const unsigned short* lds, int stride,
                                           int kt, Frag& a) {
  int lane = threadIdx.x & 31;
  int row  = lane & 15;
  int col0 = kt * 32 + ((lane >> 4) << 3);
  const unsigned short* p = lds + row * stride + col0;
  a.q[0] = *(const U128*)(p);
  a.q[1] = *(const U128*)(p + 16);
}

// B fragment from fragment-packed weights: 32 lanes x 16 bf16 contiguous per frag
__device__ __forceinline__ void load_b_packed(const unsigned short* Wp,
                                              int kt, int nt, Frag& b) {
  int lane = threadIdx.x & 31;
  const unsigned short* p = Wp + (((size_t)(kt * NT + nt) * 32 + lane) << 4);
  b.q[0] = *(const U128*)(p);
  b.q[1] = *(const U128*)(p + 8);
}

// ---------------- weight packing: row-major f32 [Kvalid][300] -> bf16 fragments ---
__global__ void pack_weight_kernel(const float* __restrict__ W,
                                   unsigned short* __restrict__ Wp,
                                   int Kvalid, int Ktiles) {
  int idx = blockIdx.x * blockDim.x + threadIdx.x;      // one lane-slot per thread
  int total = Ktiles * NT * 32;
  if (idx >= total) return;
  int lane = idx & 31;
  int frag = idx >> 5;
  int nt = frag % NT;
  int kt = frag / NT;
  int n  = nt * 16 + (lane & 15);
  Frag f;
  #pragma unroll
  for (int e = 0; e < 16; ++e) {
    int k = kt * 32 + ((e >> 3) << 4) + ((lane >> 4) << 3) + (e & 7);
    float v = (k < Kvalid && n < kHidden) ? W[(size_t)k * kHidden + n] : 0.0f;
    f.s[e] = f32_to_bf16(v);
  }
  unsigned short* dst = Wp + ((size_t)idx << 4);
  *(U128*)(dst)     = f.q[0];
  *(U128*)(dst + 8) = f.q[1];
}

__global__ void zero_kernel(float* __restrict__ p, int n) {
  int i = blockIdx.x * blockDim.x + threadIdx.x;
  if (i < n) p[i] = 0.0f;
}

// ---------------- input layer: pre = concat(atom[b2a], bond) @ W_i ; msg = relu(pre)
__global__ __launch_bounds__(256) void input_gemm_kernel(
    const float* __restrict__ atom_feats, const float* __restrict__ bond_feats,
    const int* __restrict__ b2a, const unsigned short* __restrict__ WiP,
    float* __restrict__ pre, unsigned short* __restrict__ msg) {
  __shared__ __align__(16) unsigned short sIn[16 * S_IN];
  __shared__ int sSrc[16];
  const int row0 = blockIdx.x * 16;
  if (threadIdx.x < 16) sSrc[threadIdx.x] = b2a[row0 + threadIdx.x];
  __syncthreads();
  for (int i = threadIdx.x; i < 16 * KIN; i += 256) {
    int r = i / KIN, c = i - r * KIN;
    float v = 0.0f;
    if (c < kAtomF)               v = atom_feats[(size_t)sSrc[r] * kAtomF + c];
    else if (c < kAtomF + kBondF) v = bond_feats[(size_t)(row0 + r) * kBondF + (c - kAtomF)];
    sIn[r * S_IN + c] = f32_to_bf16(v);
  }
  __syncthreads();
  const int wave = threadIdx.x >> 5;
  const int lane = threadIdx.x & 31;
  for (int nt = wave; nt < NT; nt += 8) {
    f32x8 acc = {};
    #pragma unroll
    for (int kt = 0; kt < KIN_KT; ++kt) {
      Frag a, b;
      load_a_lds(sIn, S_IN, kt, a);
      load_b_packed(WiP, kt, nt, b);
      acc = wmma_bf16(a, b, acc);
    }
    const int col = nt * 16 + (lane & 15);
    #pragma unroll
    for (int r = 0; r < 8; ++r) {
      size_t row = (size_t)(row0 + r + ((lane >> 4) << 3));
      float v = acc[r];
      pre[row * HP + col] = v;
      msg[row * HP + col] = f32_to_bf16(v > 0.0f ? v : 0.0f);
    }
  }
}

// ---------------- one message-passing step: msgOut = relu(gather(msgIn) @ W_h + pre)
__global__ __launch_bounds__(256) void mp_step_kernel(
    const unsigned short* __restrict__ msgIn, unsigned short* __restrict__ msgOut,
    const float* __restrict__ pre, const int* __restrict__ a2b,
    const int* __restrict__ b2a, const int* __restrict__ b2revb,
    const unsigned short* __restrict__ WhP) {
  __shared__ __align__(16) unsigned short sNei[16 * S_NEI];
  __shared__ int sNbr[16][kMaxNb];
  const int row0 = blockIdx.x * 16;
  if (threadIdx.x < 16 * kMaxNb) {
    int r = threadIdx.x / kMaxNb, j = threadIdx.x - r * kMaxNb;
    int bond = row0 + r;
    int nb   = a2b[(size_t)b2a[bond] * kMaxNb + j];
    int rev  = b2revb[bond];
    sNbr[r][j] = (nb >= 0 && nb != rev) ? nb : -1;
  }
  __syncthreads();
  // masked 6-neighbor sum, 8 columns at a time (b128 streaming loads)
  for (int i = threadIdx.x; i < 16 * (HP / 8); i += 256) {
    int r = i / (HP / 8), g = i - r * (HP / 8);
    int c = g * 8;
    float s0 = 0, s1 = 0, s2 = 0, s3 = 0, s4 = 0, s5 = 0, s6 = 0, s7 = 0;
    #pragma unroll
    for (int j = 0; j < kMaxNb; ++j) {
      int nb = sNbr[r][j];
      if (nb >= 0) {
        B128 m;
        m.u = *(const U128*)(msgIn + (size_t)nb * HP + c);
        s0 += bf16_to_f32(m.s[0]); s1 += bf16_to_f32(m.s[1]);
        s2 += bf16_to_f32(m.s[2]); s3 += bf16_to_f32(m.s[3]);
        s4 += bf16_to_f32(m.s[4]); s5 += bf16_to_f32(m.s[5]);
        s6 += bf16_to_f32(m.s[6]); s7 += bf16_to_f32(m.s[7]);
      }
    }
    B128 o;
    o.s[0] = f32_to_bf16(s0); o.s[1] = f32_to_bf16(s1);
    o.s[2] = f32_to_bf16(s2); o.s[3] = f32_to_bf16(s3);
    o.s[4] = f32_to_bf16(s4); o.s[5] = f32_to_bf16(s5);
    o.s[6] = f32_to_bf16(s6); o.s[7] = f32_to_bf16(s7);
    *(U128*)(sNei + r * S_NEI + c) = o.u;
  }
  __syncthreads();
  const int wave = threadIdx.x >> 5;
  const int lane = threadIdx.x & 31;
  for (int nt = wave; nt < NT; nt += 8) {
    f32x8 acc = {};
    #pragma unroll
    for (int kt = 0; kt < KH_KT; ++kt) {
      Frag a, b;
      load_a_lds(sNei, S_NEI, kt, a);
      load_b_packed(WhP, kt, nt, b);
      if (kt + 1 < KH_KT)
        __builtin_prefetch(WhP + (((size_t)((kt + 1) * NT + nt) * 32) << 4), 0, 3);
      acc = wmma_bf16(a, b, acc);
    }
    const int col = nt * 16 + (lane & 15);
    #pragma unroll
    for (int r = 0; r < 8; ++r) {
      size_t row = (size_t)(row0 + r + ((lane >> 4) << 3));
      float v = acc[r] + pre[row * HP + col];
      msgOut[row * HP + col] = f32_to_bf16(v > 0.0f ? v : 0.0f);
    }
  }
}

// ---------------- readout: atom_out = relu(concat(atom_feats, pool(msg)) @ W_o + b_o)
//                  then atomic segment-sum into pooled / counts
__global__ __launch_bounds__(256) void output_gemm_kernel(
    const float* __restrict__ atom_feats, const unsigned short* __restrict__ msg,
    const int* __restrict__ a2b, const unsigned short* __restrict__ WoP,
    const float* __restrict__ b_o, const int* __restrict__ batch,
    float* __restrict__ pooled, float* __restrict__ counts) {
  __shared__ __align__(16) unsigned short sIn[16 * S_OUT];
  __shared__ int sNbr[16][kMaxNb];
  const int row0 = blockIdx.x * 16;
  if (threadIdx.x < 16 * kMaxNb) {
    int r = threadIdx.x / kMaxNb, j = threadIdx.x - r * kMaxNb;
    sNbr[r][j] = a2b[(size_t)(row0 + r) * kMaxNb + j];
  }
  if (threadIdx.x < 16)
    atomicAdd(&counts[batch[row0 + threadIdx.x]], 1.0f);
  __syncthreads();
  for (int i = threadIdx.x; i < 16 * KOUT; i += 256) {
    int r = i / KOUT, c = i - r * KOUT;
    float v = 0.0f;
    if (c < kAtomF) {
      v = atom_feats[(size_t)(row0 + r) * kAtomF + c];
    } else if (c < kAtomF + kHidden) {
      int h = c - kAtomF;
      float s = 0.0f;
      #pragma unroll
      for (int j = 0; j < kMaxNb; ++j) {
        int nb = sNbr[r][j];
        if (nb >= 0) s += bf16_to_f32(msg[(size_t)nb * HP + h]);
      }
      v = s;
    }
    sIn[r * S_OUT + c] = f32_to_bf16(v);
  }
  __syncthreads();
  const int wave = threadIdx.x >> 5;
  const int lane = threadIdx.x & 31;
  for (int nt = wave; nt < NT; nt += 8) {
    f32x8 acc = {};
    #pragma unroll
    for (int kt = 0; kt < KOUT_KT; ++kt) {
      Frag a, b;
      load_a_lds(sIn, S_OUT, kt, a);
      load_b_packed(WoP, kt, nt, b);
      acc = wmma_bf16(a, b, acc);
    }
    const int col = nt * 16 + (lane & 15);
    if (col < kHidden) {
      float bias = b_o[col];
      #pragma unroll
      for (int r = 0; r < 8; ++r) {
        int atom = row0 + r + ((lane >> 4) << 3);
        float v = acc[r] + bias;
        v = v > 0.0f ? v : 0.0f;
        atomicAdd(&pooled[(size_t)batch[atom] * kHidden + col], v);
      }
    }
  }
}

__global__ void finalize_kernel(const float* __restrict__ pooled,
                                const float* __restrict__ counts,
                                float* __restrict__ out) {
  int i = blockIdx.x * blockDim.x + threadIdx.x;
  if (i < kGraphs * kHidden) {
    int g = i / kHidden;
    out[i] = pooled[i] / fmaxf(counts[g], 1.0f);
  }
}

// ---------------- workspace layout (bytes, all offsets multiple of 256) ----------
constexpr size_t OFF_PRE  = 0;                                        // f32 [kBonds][HP]
constexpr size_t SZ_PRE   = (size_t)kBonds * HP * 4;                  // 256,000,000
constexpr size_t OFF_MSGA = OFF_PRE + SZ_PRE;                         // bf16 [kBonds][HP]
constexpr size_t SZ_MSG   = (size_t)kBonds * HP * 2;                  // 128,000,000
constexpr size_t OFF_MSGB = OFF_MSGA + SZ_MSG;
constexpr size_t OFF_WI   = OFF_MSGB + SZ_MSG;
constexpr size_t SZ_WI    = (size_t)KIN_KT * NT * 32 * 16 * 2;        // 102,400
constexpr size_t OFF_WH   = OFF_WI + SZ_WI;
constexpr size_t SZ_WH    = (size_t)KH_KT * NT * 32 * 16 * 2;         // 204,800
constexpr size_t OFF_WO   = OFF_WH + SZ_WH;
constexpr size_t SZ_WO    = (size_t)KOUT_KT * NT * 32 * 16 * 2;       // 286,720
constexpr size_t OFF_POOL = OFF_WO + SZ_WO;
constexpr size_t SZ_POOL  = (size_t)kGraphs * kHidden * 4;            // 4,915,200
constexpr size_t OFF_CNT  = OFF_POOL + SZ_POOL;

extern "C" void kernel_launch(void* const* d_in, const int* in_sizes, int n_in,
                              void* d_out, int out_size, void* d_ws, size_t ws_size,
                              hipStream_t stream) {
  const float* atom_feats = (const float*)d_in[0];
  const float* bond_feats = (const float*)d_in[1];
  const int*   a2b        = (const int*)d_in[2];
  const int*   b2a        = (const int*)d_in[3];
  const int*   b2revb     = (const int*)d_in[4];
  const int*   batch      = (const int*)d_in[5];
  const float* W_i        = (const float*)d_in[6];
  const float* W_h        = (const float*)d_in[7];
  const float* W_o        = (const float*)d_in[8];
  const float* b_o        = (const float*)d_in[9];
  float* out = (float*)d_out;

  char* ws = (char*)d_ws;
  float*          pre   = (float*)(ws + OFF_PRE);
  unsigned short* msgA  = (unsigned short*)(ws + OFF_MSGA);
  unsigned short* msgB  = (unsigned short*)(ws + OFF_MSGB);
  unsigned short* WiP   = (unsigned short*)(ws + OFF_WI);
  unsigned short* WhP   = (unsigned short*)(ws + OFF_WH);
  unsigned short* WoP   = (unsigned short*)(ws + OFF_WO);
  float*          pool  = (float*)(ws + OFF_POOL);
  float*          cnts  = (float*)(ws + OFF_CNT);

  // 1) pack weights into WMMA fragment order (bf16, zero padded)
  {
    int tWi = KIN_KT * NT * 32, tWh = KH_KT * NT * 32, tWo = KOUT_KT * NT * 32;
    pack_weight_kernel<<<(tWi + 255) / 256, 256, 0, stream>>>(W_i, WiP, kAtomF + kBondF, KIN_KT);
    pack_weight_kernel<<<(tWh + 255) / 256, 256, 0, stream>>>(W_h, WhP, kHidden, KH_KT);
    pack_weight_kernel<<<(tWo + 255) / 256, 256, 0, stream>>>(W_o, WoP, kAtomF + kHidden, KOUT_KT);
  }
  // 2) zero pooled accumulators + counts
  {
    int nz = kGraphs * kHidden + kGraphs;
    zero_kernel<<<(nz + 255) / 256, 256, 0, stream>>>(pool, nz);
  }
  // 3) input layer
  input_gemm_kernel<<<kBonds / 16, 256, 0, stream>>>(atom_feats, bond_feats, b2a,
                                                     WiP, pre, msgA);
  // 4) DEPTH-1 message passing steps (ping-pong)
  unsigned short* cur = msgA;
  unsigned short* nxt = msgB;
  for (int d = 0; d < kDepth - 1; ++d) {
    mp_step_kernel<<<kBonds / 16, 256, 0, stream>>>(cur, nxt, pre, a2b, b2a, b2revb, WhP);
    unsigned short* t = cur; cur = nxt; nxt = t;
  }
  // 5) readout + atomic segment sums
  output_gemm_kernel<<<kAtoms / 16, 256, 0, stream>>>(atom_feats, cur, a2b, WoP,
                                                      b_o, batch, pool, cnts);
  // 6) mean per graph
  finalize_kernel<<<(kGraphs * kHidden + 255) / 256, 256, 0, stream>>>(pool, cnts, out);
}
